// EnvGenerator_55018531062592
// MI455X (gfx1250) — matrix operand order repaired and checked by reading, not compiled, not run on hardware
//
#include <hip/hip_runtime.h>
#include <math.h>

// ---------------------------------------------------------------------------
// K=8 edge MLPs (384 -> 256 relu -> 1) + Gumbel-sigmoid gates.
// 315 GFLOP batched GEMM dominates => v_wmma_f32_16x16x32_bf16, f32 accum.
// W1 is pre-converted/pre-transposed to bf16 in d_ws by a tiny prep kernel,
// so per-k-step staging is 4x contiguous 16B global_load_async_to_lds_b128
// copies per thread (ASYNCcnt), double-buffered against the WMMAs.
// A tiles convert fp32->bf16 on the fly (v_cvt via __builtin_convertvector).
// Expert-adjacent grid -> A tiles reused out of the 192MB L2 (HBM ~ 1 pass).
// ---------------------------------------------------------------------------

#define NEXPERT 8
#define IN_FEAT 384
#define MLP_DIM 256
#define BIAS_C  1e-4f

#define MTILE 64        // edge rows per workgroup
#define KC    128       // A K-chunk in LDS (3 chunks x 4 wmma k-steps)
#define NSTEP 12        // total wmma k-steps (384/32)
#define AS_STRIDE 136   // 128 + 8 pad -> 272B row, 16B aligned
#define BT_STRIDE 40    // 32 + 8 pad  ->  80B row, 16B aligned

typedef __bf16 bf16x4  __attribute__((ext_vector_type(4)));
typedef __bf16 bf16x8  __attribute__((ext_vector_type(8)));
typedef __bf16 bf16x16 __attribute__((ext_vector_type(16)));
typedef float  f32x4   __attribute__((ext_vector_type(4)));
typedef float  f32x8   __attribute__((ext_vector_type(8)));

union V16 { bf16x16 v; bf16x8 h[2]; };

struct __align__(16) SMem {
  __bf16 As[MTILE * AS_STRIDE];        // A chunk              (17408 B)
  __bf16 Bt[2][MLP_DIM * BT_STRIDE];   // W1 k-step, transposed, x2 (40960 B)
  float  logits[MTILE];                // (256 B)   total 58624 B
};

// ---- prep: W1 [8,384,256] f32 -> ws bf16, layout [k*12+g][n][kr(0..31)] ----
// note (k*384 + g*32)*256 == (k*12+g)*32*256, so blockIdx.x = k*12+g indexes
// both source rows and destination block directly.
__global__ __launch_bounds__(256)
void prep_w1(const float* __restrict__ W1, __bf16* __restrict__ wsb) {
  const int kg = blockIdx.x;                 // 0..95
  const int n  = threadIdx.x;                // 0..255
  const float* src = W1  + (size_t)kg * 32 * MLP_DIM;
  __bf16*      dst = wsb + (size_t)kg * MLP_DIM * 32;
#pragma unroll
  for (int kr = 0; kr < 32; ++kr)
    dst[n * 32 + kr] = (__bf16)src[kr * MLP_DIM + n];
}

__global__ __launch_bounds__(256)
void edge_mlp_gumbel(const float* __restrict__ TE,     // [E,384]
                     const float* __restrict__ noise,  // [8,E]
                     const float* __restrict__ W1,     // [8,384,256] (fallback)
                     const float* __restrict__ b1,     // [8,256]
                     const float* __restrict__ W2,     // [8,256]
                     const float* __restrict__ b2,     // [8]
                     const __bf16* __restrict__ wsb,   // prepped W1 (or null)
                     float* __restrict__ out,          // [2,8,E]
                     int E, int useWs)
{
  __shared__ SMem sm;
  const int tid  = threadIdx.x;
  const int kexp = blockIdx.x & 7;   // experts adjacent -> A tile L2 reuse
  const int mt   = blockIdx.x >> 3;
  const int e0   = mt * MTILE;

  const int wave  = tid >> 5;
  const int lane  = tid & 31;
  const int lhalf = lane >> 4;         // 0/1
  const int l16   = lane & 15;
  const int r0    = (wave & 3) * 16;   // 16-row block within the 64-row tile
  const int c0    = (wave >> 2) * 128; // 128-col half of the 256 hidden cols

  if (tid < MTILE) sm.logits[tid] = 0.0f;

  f32x8 acc[8];
  const f32x8 zero = {0.f,0.f,0.f,0.f,0.f,0.f,0.f,0.f};
#pragma unroll
  for (int t = 0; t < 8; ++t) acc[t] = zero;

  // ---- stage one W1 k-step (32x256) into Bt[dbuf] ----
  auto stageB = [&](int gstep, int dbuf) {
    if (useWs) {
      // prepped bf16, already transposed: 4 contiguous 16B async copies/thread
      const __bf16* src = wsb + (size_t)(kexp * NSTEP + gstep) * MLP_DIM * 32;
#pragma unroll
      for (int j = 0; j < 4; ++j) {
        int i4 = tid + 256 * j;                 // 16B chunk id: n*4 + c
        int n = i4 >> 2, c = i4 & 3;
        unsigned lds = (unsigned)(unsigned long long)
            &sm.Bt[dbuf][n * BT_STRIDE + c * 8];
        unsigned long long ga = (unsigned long long)(src + (size_t)i4 * 8);
        asm volatile("global_load_async_to_lds_b128 %0, %1, off"
                     :: "v"(lds), "v"(ga) : "memory");
      }
    } else {
      // fallback: convert+transpose from f32 W1
      const float* src = W1 + ((size_t)kexp * IN_FEAT + gstep * 32) * MLP_DIM;
#pragma unroll 4
      for (int j = 0; j < 32; ++j) {
        int i = tid + 256 * j;
        int kr = i >> 8, nn = i & 255;
        sm.Bt[dbuf][nn * BT_STRIDE + kr] = (__bf16)src[kr * MLP_DIM + nn];
      }
    }
  };

  // ---- prologue: stage k-step 0 into Bt[0] ----
  stageB(0, 0);

  for (int g = 0; g < NSTEP; ++g) {
    const int buf = g & 1;
    const int ksl = (g & 3) * 32;      // k offset within current A chunk

    if ((g & 3) == 0) {
      // new A chunk: wait until every wave finished reading the old one
      asm volatile("s_wait_asynccnt 0x0" ::: "memory");
      __syncthreads();
      const int chunk = g >> 2;
      // stage A chunk [64 x 128] fp32 -> bf16 (packed cvt), b64 stores
#pragma unroll
      for (int j = 0; j < 8; ++j) {
        int i4 = tid + 256 * j;
        int r = i4 >> 5, c4 = i4 & 31;
        f32x4 v = *reinterpret_cast<const f32x4*>(
            TE + (size_t)(e0 + r) * IN_FEAT + chunk * KC + c4 * 4);
        bf16x4 h = __builtin_convertvector(v, bf16x4);
        *reinterpret_cast<bf16x4*>(&sm.As[r * AS_STRIDE + c4 * 4]) = h;
      }
    }

    // Publish Bt[buf] staged last step (and As if just staged); also ensures
    // the buffer staged below (Bt[!buf], last read in step g-1) is free.
    asm volatile("s_wait_asynccnt 0x0" ::: "memory");
    __syncthreads();

    // stage next W1 k-step into the other buffer (overlaps the WMMAs)
    if (g + 1 < NSTEP) stageB(g + 1, buf ^ 1);

    // ---- A fragment (16x32 bf16): lane = row l16, K = 8*lhalf+[0..7], +16 ----
    V16 a;
    const __bf16* ap = &sm.As[(r0 + l16) * AS_STRIDE + ksl + lhalf * 8];
    a.h[0] = *reinterpret_cast<const bf16x8*>(ap);
    a.h[1] = *reinterpret_cast<const bf16x8*>(ap + 16);

    // ---- B fragments (32x16 bf16): lane = col l16, K = 8*lhalf+[0..7], +16 ----
    V16 b[8];
#pragma unroll
    for (int t = 0; t < 8; ++t) {
      const __bf16* bp =
          &sm.Bt[buf][(c0 + t * 16 + l16) * BT_STRIDE + lhalf * 8];
      b[t].h[0] = *reinterpret_cast<const bf16x8*>(bp);
      b[t].h[1] = *reinterpret_cast<const bf16x8*>(bp + 16);
    }
#pragma unroll
    for (int t = 0; t < 8; ++t) {
      acc[t] = __builtin_amdgcn_wmma_f32_16x16x32_bf16(
          false, a.v, false, b[t].v, (short)0, acc[t], false, false);
    }
  }

  // ---- fused layer 2: logits[m] = sum_n relu(hidden + b1[n]) * W2[n] ----
  // C layout: acc[t][g] is (row r0 + g + 8*lhalf, col c0 + 16t + l16)
  float partial[8];
#pragma unroll
  for (int g = 0; g < 8; ++g) partial[g] = 0.0f;
#pragma unroll
  for (int t = 0; t < 8; ++t) {
    int n = c0 + t * 16 + l16;
    float bb  = b1[kexp * MLP_DIM + n];
    float w2v = W2[kexp * MLP_DIM + n];
#pragma unroll
    for (int g = 0; g < 8; ++g) {
      float h = acc[t][g] + bb;
      h = (h > 0.0f) ? h : 0.0f;
      partial[g] += h * w2v;
    }
  }
  const int mrow = r0 + lhalf * 8;
#pragma unroll
  for (int g = 0; g < 8; ++g)
    atomicAdd(&sm.logits[mrow + g], partial[g]);   // ds_add_f32

  __syncthreads();

  // ---- Gumbel-sigmoid gate + straight-through hard threshold ----
  if (tid < MTILE) {
    const int e = e0 + tid;
    float logit = sm.logits[tid] + b2[kexp];
    float u   = noise[(size_t)kexp * E + e];
    float eps = (1.0f - BIAS_C) - (1.0f - 2.0f * BIAS_C) * u;
    float gn  = logf(eps) - log1pf(-eps);
    float gi  = gn + logit;                 // TEMPERATURE = 1
    float ew  = 1.0f / (1.0f + expf(-gi));
    const size_t KE = (size_t)NEXPERT * (size_t)E;
    out[(size_t)kexp * E + e]      = ew;                         // edge_weight
    out[KE + (size_t)kexp * E + e] = (ew >= 0.5f) ? 1.0f : 0.0f; // hard (ST)
  }
}

extern "C" void kernel_launch(void* const* d_in, const int* in_sizes, int n_in,
                              void* d_out, int out_size, void* d_ws, size_t ws_size,
                              hipStream_t stream) {
  const float* TE    = (const float*)d_in[0];  // triplet_emb [E,384]
  const float* noise = (const float*)d_in[1];  // [8,E]
  const float* W1    = (const float*)d_in[2];  // [8,384,256]
  const float* b1    = (const float*)d_in[3];  // [8,256]
  const float* W2    = (const float*)d_in[4];  // [8,256]
  const float* b2    = (const float*)d_in[5];  // [8]
  // d_in[6] = edge_index : unused by the reference computation
  float* out = (float*)d_out;

  const int E = in_sizes[0] / IN_FEAT;         // 200000 (divisible by MTILE=64)
  const int mtiles = E / MTILE;                // 3125

  const size_t wsNeed = (size_t)NEXPERT * NSTEP * MLP_DIM * 32 * 2; // 1.5 MB
  const int useWs = (d_ws != nullptr && ws_size >= wsNeed) ? 1 : 0;
  if (useWs) {
    prep_w1<<<NEXPERT * NSTEP, 256, 0, stream>>>(W1, (__bf16*)d_ws);
  }
  dim3 grid(mtiles * NEXPERT), block(256);
  edge_mlp_gumbel<<<grid, block, 0, stream>>>(
      TE, noise, W1, b1, W2, b2, (const __bf16*)d_ws, out, E, useWs);
}
